// Feedback_68367289418011
// MI455X (gfx1250) — matrix operand order repaired and checked by reading, not compiled, not run on hardware
//
#include <hip/hip_runtime.h>
#include <hip/hip_bf16.h>

// ---------------------------------------------------------------------------
// LSTM forecast (B=1024, T=512 warmup, 64 AR outputs) for MI455X / gfx1250.
//
// Persistent-workgroup design: 32 WGs x 512 threads (16 wave32s). Each WG owns
// 32 batch rows for the entire 575-step recurrence. Per step:
//   z[32,2048] = [x|h][32,576] @ Wpack[576,2048]   via v_wmma_f32_16x16x32_bf16
//   gates -> c,h update (c in VGPRs, h -> LDS bf16)
//   (AR phase) pred[32,64] = h @ Wd + b  -> d_out and fed back as next x.
// Weights are packed once to bf16 in WMMA B-fragment order so every lane's
// fragment is a contiguous 32-byte NT global load (L2-resident, WGP$-bypass).
// sched_group_barrier pins the schedule to 8-load / 8-WMMA groups per K-tile,
// forcing 4 B buffers live so L2 latency overlaps a full window of WMMAs.
// ---------------------------------------------------------------------------

typedef __attribute__((ext_vector_type(16))) __bf16       v16bf;
typedef __attribute__((ext_vector_type(8)))  float        v8f;
typedef __attribute__((ext_vector_type(4)))  unsigned int u32x4; // POD 128-bit

union FragB {
    u32x4 q[2];
    v16bf v;
};

#define T_WARM    512
#define OUT_STEPS 64
#define BM        32   // batch rows per workgroup

#if __has_builtin(__builtin_amdgcn_sched_group_barrier)
#define SCHED_GROUP(mask, size, id) __builtin_amdgcn_sched_group_barrier(mask, size, id)
#else
#define SCHED_GROUP(mask, size, id)
#endif
#define SG_MFMA  0x008   // WMMA instructions
#define SG_VMEMR 0x020   // VMEM read instructions

__device__ __forceinline__ float fast_sigmoid(float x) {
    return 1.0f / (1.0f + __expf(-x));
}
__device__ __forceinline__ float fast_tanh(float x) {
    return 2.0f / (1.0f + __expf(-2.0f * x)) - 1.0f;
}

// 32-byte non-temporal global fragment load (weights: L2-resident, WGP$-bypass)
__device__ __forceinline__ FragB load_frag_nt(const __bf16* __restrict__ p) {
    FragB f;
    f.q[0] = __builtin_nontemporal_load((const u32x4*)(p));
    f.q[1] = __builtin_nontemporal_load((const u32x4*)(p + 8));
    return f;
}
// 32-byte LDS fragment load split as two b128 (A-matrix 16x32 bf16 layout)
__device__ __forceinline__ FragB load_frag_lds(const __bf16* p) {
    FragB f;
    f.q[0] = *(const u32x4*)(p);
    f.q[1] = *(const u32x4*)(p + 8);    // second K-run at +8 elements
    return f;
}

// ---------------------------------------------------------------------------
// Pack fp32 weights -> bf16 in WMMA B-fragment tile order.
// Combined W[576][2048]: rows 0..63 = kernel (x), rows 64..575 = rec_kernel (h).
// Tile (kt,nt) covers K=[kt*32,+32), N=[nt*16,+16). Within a tile, lane l
// (n=l&15, khalf=l>>4) stores its 16 elements W[kt*32+khalf*16+j][n], j=0..15,
// contiguously at tile*512 + l*16  (so the GEMM loads 2x b128 per fragment).
// ---------------------------------------------------------------------------
__global__ void pack_weights(const float* __restrict__ kern,  // [64][2048]
                             const float* __restrict__ rec,   // [512][2048]
                             const float* __restrict__ dw,    // [512][64]
                             __bf16* __restrict__ wpack,      // 18*128 tiles
                             __bf16* __restrict__ wdpack)     // 16*4 tiles
{
    const int bid   = blockIdx.x;
    const int lane  = threadIdx.x;       // 32 threads = 1 tile
    const int nloc  = lane & 15;
    const int khalf = lane >> 4;

    if (bid < 18 * 128) {                          // LSTM weights
        const int kt = bid >> 7;
        const int nt = bid & 127;
        const int n  = nt * 16 + nloc;
        const int kb = kt * 32 + khalf * 16;
        __bf16* dst = wpack + (size_t)bid * 512 + lane * 16;
#pragma unroll
        for (int j = 0; j < 16; ++j) {
            const int k = kb + j;
            const float v = (k < 64) ? kern[k * 2048 + n]
                                     : rec[(k - 64) * 2048 + n];
            dst[j] = (__bf16)v;
        }
    } else {                                       // dense weights
        const int idx = bid - 18 * 128;            // kt*4+nt, kt<16, nt<4
        const int n   = (idx & 3) * 16 + nloc;
        const int kb  = (idx >> 2) * 32 + khalf * 16;
        __bf16* dst = wdpack + (size_t)idx * 512 + lane * 16;
#pragma unroll
        for (int j = 0; j < 16; ++j)
            dst[j] = (__bf16)dw[(kb + j) * 64 + n];
    }
}

// ---------------------------------------------------------------------------
// Persistent recurrence kernel.
// Wave w owns units u in [w*32, w*32+32): accumulator n-tiles j=0..7 map to
// column  col(j) = (j>>1)*512 + w*32 + (j&1)*16  (gate-aligned -> no z exchange).
// ---------------------------------------------------------------------------
__global__ __launch_bounds__(512, 1)
void lstm_forecast(const float* __restrict__ x,      // [1024][512][64]
                   const float* __restrict__ bias,   // [2048]
                   const float* __restrict__ dbias,  // [64]
                   const __bf16* __restrict__ wpack,
                   const __bf16* __restrict__ wdpack,
                   float* __restrict__ out)          // [1024][64][64]
{
    __shared__ __attribute__((aligned(16))) __bf16 hbuf[32][520]; // 512 + pad
    __shared__ __attribute__((aligned(16))) __bf16 xbuf[32][72];  // 64  + pad

    const int tid   = threadIdx.x;
    const int wave  = tid >> 5;            // 0..15
    const int lane  = tid & 31;
    const int nloc  = lane & 15;
    const int khalf = lane >> 4;
    const int rb    = blockIdx.x * BM;

    // h0 = 0
    for (int i = tid; i < 32 * 520; i += 512) ((__bf16*)hbuf)[i] = (__bf16)0.0f;

    // cell state in registers, D-matrix layout: c[mt][uh][v] = c[row][unit]
    float cst[2][2][8];
#pragma unroll
    for (int mt = 0; mt < 2; ++mt)
#pragma unroll
        for (int uh = 0; uh < 2; ++uh)
#pragma unroll
            for (int v = 0; v < 8; ++v) cst[mt][uh][v] = 0.0f;

    // per-column biases (same value for all 8 rows of a D tile)
    float bj[8];
#pragma unroll
    for (int j = 0; j < 8; ++j)
        bj[j] = bias[(j >> 1) * 512 + wave * 32 + (j & 1) * 16 + nloc];
    const float db = dbias[(wave & 3) * 16 + nloc];

    // invariant part of B-fragment addressing (per-wave n-tile pair + lane slot)
    const __bf16* wbase = wpack + (size_t)(wave * 2) * 512 + lane * 16;

    __syncthreads();

    for (int t = 0; t < T_WARM + OUT_STEPS - 1; ++t) {
        // ---- stage x_t (fp32 -> bf16); AR phase reuses pred left in xbuf ----
        if (t < T_WARM) {
            const int idx = tid * 4;               // 2048 elems / 512 threads
            const int r = idx >> 6, f = idx & 63;
            const u32x4 raw = __builtin_nontemporal_load(
                (const u32x4*)(x + (((size_t)(rb + r)) * T_WARM + t) * 64 + f));
            const float* xv = (const float*)&raw;
            xbuf[r][f + 0] = (__bf16)xv[0];
            xbuf[r][f + 1] = (__bf16)xv[1];
            xbuf[r][f + 2] = (__bf16)xv[2];
            xbuf[r][f + 3] = (__bf16)xv[3];
        }
        __syncthreads();

        // ---- z = [x|h] @ W + b  (18 K-tiles of 32; 2 M-tiles; 8 N-tiles) ----
        v8f acc[2][8];
#pragma unroll
        for (int mt = 0; mt < 2; ++mt)
#pragma unroll
            for (int j = 0; j < 8; ++j) {
                v8f a;
#pragma unroll
                for (int e = 0; e < 8; ++e) a[e] = bj[j];
                acc[mt][j] = a;
            }

#pragma unroll 1
        for (int kt = 0; kt < 18; ++kt) {
            FragB af[2];
            if (kt < 2) {                         // x features, K = kt*32..
                const int col = kt * 32 + khalf * 16;
#pragma unroll
                for (int mt = 0; mt < 2; ++mt)
                    af[mt] = load_frag_lds(&xbuf[mt * 16 + nloc][col]);
            } else {                              // h units, K = (kt-2)*32..
                const int col = (kt - 2) * 32 + khalf * 16;
#pragma unroll
                for (int mt = 0; mt < 2; ++mt)
                    af[mt] = load_frag_lds(&hbuf[mt * 16 + nloc][col]);
            }
            const __bf16* wkt = wbase + (size_t)kt * 128 * 512;
            // two half-sweeps of 4 B fragments each; sched_group_barrier pins
            // 8 loads ahead of their 8 WMMAs (4 live buffers, staged waits).
#pragma unroll
            for (int gg = 0; gg < 2; ++gg) {
                FragB b[4];
#pragma unroll
                for (int q = 0; q < 4; ++q) {
                    const int g = gg * 2 + (q >> 1);   // gate index 0..3
                    const int u = q & 1;               // unit-halve
                    b[q] = load_frag_nt(wkt + ((size_t)g * 32 + u) * 512);
                }
#pragma unroll
                for (int q = 0; q < 4; ++q) {
                    const int j = gg * 4 + q;          // acc n-tile index
                    acc[0][j] = __builtin_amdgcn_wmma_f32_16x16x32_bf16(
                        false, af[0].v, false, b[q].v, (short)0, acc[0][j],
                        false, false);
                    acc[1][j] = __builtin_amdgcn_wmma_f32_16x16x32_bf16(
                        false, af[1].v, false, b[q].v, (short)0, acc[1][j],
                        false, false);
                }
            }
            // Pin the pipeline for this K-tile's basic block:
            //   8 vmem loads -> 8 wmma -> 8 vmem loads -> 8 wmma
            SCHED_GROUP(SG_VMEMR, 8, 0);
            SCHED_GROUP(SG_MFMA,  8, 0);
            SCHED_GROUP(SG_VMEMR, 8, 0);
            SCHED_GROUP(SG_MFMA,  8, 0);
        }
        __syncthreads();   // all hbuf/xbuf reads done before h rewrite

        // ---- gates -> c,h ; write h (bf16) back to LDS in row-major ----
#pragma unroll
        for (int mt = 0; mt < 2; ++mt)
#pragma unroll
            for (int uh = 0; uh < 2; ++uh)
#pragma unroll
                for (int v = 0; v < 8; ++v) {
                    const float zi = acc[mt][0 + uh][v];
                    const float zf = acc[mt][2 + uh][v];
                    const float zg = acc[mt][4 + uh][v];
                    const float zo = acc[mt][6 + uh][v];
                    const float cn = fast_sigmoid(zf) * cst[mt][uh][v] +
                                     fast_sigmoid(zi) * fast_tanh(zg);
                    cst[mt][uh][v] = cn;
                    const float hv = fast_sigmoid(zo) * fast_tanh(cn);
                    const int m = mt * 16 + v + (khalf << 3);
                    hbuf[m][wave * 32 + uh * 16 + nloc] = (__bf16)hv;
                }
        __syncthreads();

        // ---- dense projection: pred = h @ Wd + b ; emit output + feedback ----
        if (t >= T_WARM - 1) {
            if (wave < 8) {                       // 2 M-tiles x 4 N-tiles
                const int mt = wave >> 2, nt = wave & 3;
                v8f pacc;
#pragma unroll
                for (int e = 0; e < 8; ++e) pacc[e] = db;
#pragma unroll 2
                for (int kt = 0; kt < 16; ++kt) {
                    const FragB af =
                        load_frag_lds(&hbuf[mt * 16 + nloc][kt * 32 + khalf * 16]);
                    const FragB bf = load_frag_nt(
                        wdpack + ((size_t)(kt * 4 + nt)) * 512 + lane * 16);
                    pacc = __builtin_amdgcn_wmma_f32_16x16x32_bf16(
                        false, af.v, false, bf.v, (short)0, pacc, false, false);
                }
                const int s = t - (T_WARM - 1);   // 0..63
                const int f = nt * 16 + nloc;
#pragma unroll
                for (int v = 0; v < 8; ++v) {
                    const int m = mt * 16 + v + (khalf << 3);
                    __builtin_nontemporal_store(
                        pacc[v],
                        out + (((size_t)(rb + m)) * OUT_STEPS + s) * 64 + f);
                    xbuf[m][f] = (__bf16)pacc[v];  // feedback as next x_t
                }
            }
            __syncthreads();
        }
    }
}

extern "C" void kernel_launch(void* const* d_in, const int* in_sizes, int n_in,
                              void* d_out, int out_size, void* d_ws, size_t ws_size,
                              hipStream_t stream) {
    const float* xin   = (const float*)d_in[0];  // [1024,512,64]
    const float* kern  = (const float*)d_in[1];  // [64,2048]
    const float* rec   = (const float*)d_in[2];  // [512,2048]
    const float* bias  = (const float*)d_in[3];  // [2048]
    const float* dw    = (const float*)d_in[4];  // [512,64]
    const float* dbias = (const float*)d_in[5];  // [64]

    __bf16* wpack  = (__bf16*)d_ws;                       // 18*128*512 bf16
    __bf16* wdpack = wpack + (size_t)18 * 128 * 512;      // + 64*512 bf16

    pack_weights<<<18 * 128 + 64, 32, 0, stream>>>(kern, rec, dw, wpack, wdpack);
    lstm_forecast<<<1024 / BM, 512, 0, stream>>>(xin, bias, dbias, wpack, wdpack,
                                                 (float*)d_out);
}
